// MultiHeadAttentionEINSUM_60739427500256
// MI455X (gfx1250) — matrix operand order repaired and checked by reading, not compiled, not run on hardware
//
#include <hip/hip_runtime.h>
#include <hip/hip_bf16.h>

#define DI 1024
#define DOUT 1024
#define NBATCH 4
#define NSEQ 2048
#define NHEAD 16
#define HDIM 64

typedef __attribute__((ext_vector_type(16))) __bf16 bf16x16;
typedef __attribute__((ext_vector_type(8)))  __bf16 bf16x8;
typedef __attribute__((ext_vector_type(8)))  float  f32x8;

union BFrag { bf16x16 f; bf16x8 h[2]; };

// ---- CDNA5 async global->LDS path (probe-verified on this toolchain) -----
#if defined(__HIP_DEVICE_COMPILE__) && defined(__has_builtin)
#if __has_builtin(__builtin_amdgcn_global_load_async_to_lds_b128) && \
    __has_builtin(__builtin_amdgcn_s_wait_asynccnt)
#define HAVE_ASYNC_LDS 1
#endif
#endif
#ifndef HAVE_ASYNC_LDS
#define HAVE_ASYNC_LDS 0
#endif

#if HAVE_ASYNC_LDS
typedef __attribute__((__vector_size__(4 * sizeof(int)))) int i32x4v;
__device__ __forceinline__ void cp16_async(unsigned short* lds, const unsigned short* g) {
  __builtin_amdgcn_global_load_async_to_lds_b128(
      (__attribute__((address_space(1))) i32x4v*)(g),
      (__attribute__((address_space(3))) i32x4v*)(lds), 0, 0);
}
__device__ __forceinline__ void async_wait0() { __builtin_amdgcn_s_wait_asynccnt(0); }
#else
__device__ __forceinline__ void cp16_async(unsigned short* lds, const unsigned short* g) {
  *(uint4*)lds = *(const uint4*)g;   // sync fallback: global load + ds store
}
__device__ __forceinline__ void async_wait0() {}
#endif

__device__ __forceinline__ unsigned short f2bf(float x) {
  unsigned int u = __float_as_uint(x);
  u += 0x7fffu + ((u >> 16) & 1u);          // round-to-nearest-even
  return (unsigned short)(u >> 16);
}

// ---------------------------------------------------------------- converts
__global__ void cvt_bf16(const float* __restrict__ s,
                         unsigned short* __restrict__ d, int n) {
  int i = blockIdx.x * blockDim.x + threadIdx.x;
  int stride = gridDim.x * blockDim.x;
  for (; i < n; i += stride) d[i] = f2bf(s[i]);
}

// d[n*1024+k] = bf16(s[k*1024+n])  (one-time weight transpose, 1024x1024)
__global__ void __launch_bounds__(256) cvt_bf16_t(const float* __restrict__ s,
                                                  unsigned short* __restrict__ d) {
  __shared__ float tile[32][33];
  const int bx = blockIdx.x * 32, by = blockIdx.y * 32;
  const int tx = threadIdx.x, ty = threadIdx.y;   // block (32, 8)
#pragma unroll
  for (int i = 0; i < 32; i += 8)
    tile[ty + i][tx] = s[(size_t)(by + ty + i) * DOUT + bx + tx];
  __syncthreads();
#pragma unroll
  for (int i = 0; i < 32; i += 8)
    d[(size_t)(bx + ty + i) * DI + by + tx] = f2bf(tile[tx][ty + i]);
}

// ------------------------------------------------------- shared GEMM core
// acc[8] (16x128 per wave) += A[128x1024 rows mbase..] * Bt[nbase..][k]^T
// Bt is [n][k] row-major so staging is a contiguous b128 async copy.
// Double-buffered LDS, one barrier per k-step, B-frags preloaded in bulk.
__device__ __forceinline__ void gemm_mainloop_bf16(
    const unsigned short* __restrict__ A,
    const unsigned short* __restrict__ Bt,
    int mbase, int nbase,
    unsigned short* As, unsigned short* Bs,   // each 2*128*32 elems
    f32x8 acc[8]) {
  const int t = threadIdx.x;
  const int w = t >> 5, lane = t & 31, hl = lane >> 4, ln = lane & 15;
  const int srow = t >> 1, sc0 = (t & 1) * 16;

  const unsigned short* aSrc = A  + (size_t)(mbase + srow) * DI + sc0;
  const unsigned short* bSrc = Bt + (size_t)(nbase + srow) * DI + sc0;

  // prologue: stage k-tile 0 into buffer 0
  cp16_async(As + srow * 32 + sc0,     aSrc);
  cp16_async(As + srow * 32 + sc0 + 8, aSrc + 8);
  cp16_async(Bs + srow * 32 + sc0,     bSrc);
  cp16_async(Bs + srow * 32 + sc0 + 8, bSrc + 8);

  for (int ki = 0; ki < DI / 32; ++ki) {
    const int cur = ki & 1;
    async_wait0();
    __syncthreads();
    if (ki + 1 < DI / 32) {                       // prefetch next tile
      const int nxt = cur ^ 1;
      const unsigned short* as2 = aSrc + (ki + 1) * 32;
      const unsigned short* bs2 = bSrc + (ki + 1) * 32;
      unsigned short* ad2 = As + nxt * 4096 + srow * 32 + sc0;
      unsigned short* bd2 = Bs + nxt * 4096 + srow * 32 + sc0;
      cp16_async(ad2,     as2);  cp16_async(ad2 + 8, as2 + 8);
      cp16_async(bd2,     bs2);  cp16_async(bd2 + 8, bs2 + 8);
    }
    const unsigned short* Ab = As + cur * 4096;
    const unsigned short* Bb = Bs + cur * 4096;
    BFrag af;   // A-layout: elem j -> K = (j<8 ? 8*hl+j : 16+8*hl+(j-8))
    af.h[0] = *(const bf16x8*)(Ab + (w * 16 + ln) * 32 + 8 * hl);
    af.h[1] = *(const bf16x8*)(Ab + (w * 16 + ln) * 32 + 8 * hl + 16);
    BFrag bf[8];  // preload ALL B-frags so WMMAs can issue back-to-back
#pragma unroll
    for (int c = 0; c < 8; ++c) {
      const unsigned short* bp = Bb + (c * 16 + ln) * 32 + hl * 8;
      bf[c].h[0] = *(const bf16x8*)bp;
      bf[c].h[1] = *(const bf16x8*)(bp + 16);
    }
#pragma unroll
    for (int c = 0; c < 8; ++c)
      acc[c] = __builtin_amdgcn_wmma_f32_16x16x32_bf16(
          false, af.f, false, bf[c].f, (short)0, acc[c], false, false);
  }
}

// ---------------------------------------------------------------- QKV GEMM
__global__ void __launch_bounds__(256, 1) qkv_gemm(
    const unsigned short* __restrict__ xb,
    const unsigned short* __restrict__ wqT,
    const unsigned short* __restrict__ wkT,
    const unsigned short* __restrict__ wvT,
    unsigned short* __restrict__ Qo,
    unsigned short* __restrict__ Ko,
    unsigned short* __restrict__ Vo) {
  __shared__ unsigned short As[2 * 128 * 32];
  __shared__ unsigned short Bs[2 * 128 * 32];

  const int z = blockIdx.z;
  const unsigned short* wb = (z == 0) ? wqT : (z == 1) ? wkT : wvT;
  unsigned short* dst = (z == 0) ? Qo : (z == 1) ? Ko : Vo;
  const float scale = (z == 0) ? 0.125f : 1.0f;   // fold 1/sqrt(64) into Q
  const int nbase = blockIdx.x * 128, mbase = blockIdx.y * 128;

  f32x8 acc[8] = {};
  gemm_mainloop_bf16(xb, wb, mbase, nbase, As, Bs, acc);

  const int t = threadIdx.x;
  const int w = t >> 5, lane = t & 31, hl = lane >> 4, ln = lane & 15;
#pragma unroll
  for (int c = 0; c < 8; ++c) {
    const int col = nbase + c * 16 + ln;
    const int h = col >> 6, d = col & 63;
#pragma unroll
    for (int e = 0; e < 8; ++e) {
      const int m = mbase + w * 16 + e + 8 * hl;
      const int b = m >> 11, n = m & 2047;
      const unsigned short v = f2bf(acc[c][e] * scale);
      if (z == 2)  // V stored head-transposed [b][h][d][n] for attention staging
        dst[((size_t)((b * NHEAD + h) * HDIM + d)) * NSEQ + n] = v;
      else         // Q, K stored [b][h][n][d]
        dst[((size_t)((b * NHEAD + h) * NSEQ + n) << 6) + d] = v;
    }
  }
}

// ---------------------------------------------------------------- attention
// grid (16 qtiles, 64 bh), 8 waves/block; wave owns 16 query rows.
__global__ void __launch_bounds__(256, 1) attn_kernel(
    const unsigned short* __restrict__ Qh,
    const unsigned short* __restrict__ Kh,
    const unsigned short* __restrict__ Vt,   // [b][h][d][n]
    unsigned short* __restrict__ ctx) {
  __shared__ unsigned short Vs[2 * 64 * 32];    // double-buffered [d][key] chunk
  __shared__ unsigned short Ps[8 * 16 * 32];    // per-wave P tile 16x32

  const int qtile = blockIdx.x;
  const int bh = blockIdx.y;
  const int b = bh >> 4, h = bh & 15;
  const int qbase = qtile * 128;

  const int t = threadIdx.x;
  const int w = t >> 5, lane = t & 31, hl = lane >> 4, ln = lane & 15;

  const unsigned short* Qp = Qh + (size_t)bh * NSEQ * HDIM;
  const unsigned short* Kp = Kh + (size_t)bh * NSEQ * HDIM;
  const unsigned short* Vp = Vt + (size_t)bh * NSEQ * HDIM;

  BFrag qf[2];  // Q A-fragments, d = s*32 + (A-layout K)
  {
    const int qrow = qbase + w * 16 + ln;
#pragma unroll
    for (int s = 0; s < 2; ++s) {
      const unsigned short* p = Qp + (size_t)qrow * HDIM + s * 32 + 8 * hl;
      qf[s].h[0] = *(const bf16x8*)p;
      qf[s].h[1] = *(const bf16x8*)(p + 16);
    }
  }

  f32x8 O[4] = {};
  float mrow[8], lrow[8];
#pragma unroll
  for (int e = 0; e < 8; ++e) { mrow[e] = -1e30f; lrow[e] = 0.0f; }

  const int cmax = qtile * 4 + 4;
  const int qwave_hi = qbase + w * 16 + 15;
  const int vd = t >> 2, vk0 = (t & 3) * 8;     // V staging map (contiguous)
  const unsigned short* vSrc = Vp + (size_t)vd * NSEQ + vk0;

  cp16_async(Vs + vd * 32 + vk0, vSrc);         // prologue: chunk 0 -> buf 0

  for (int c = 0; c < cmax; ++c) {
    const int cur = c & 1;
    async_wait0();
    __syncthreads();
    if (c + 1 < cmax)                           // prefetch next V chunk
      cp16_async(Vs + (cur ^ 1) * 2048 + vd * 32 + vk0, vSrc + (c + 1) * 32);

    if (c * 32 <= qwave_hi) {   // wave-uniform causal skip
      f32x8 s0 = {}, s1 = {};
#pragma unroll
      for (int s = 0; s < 2; ++s) {   // K-fragments direct from global (d-contig)
        BFrag k0, k1;
        const unsigned short* p0 = Kp + (size_t)(c * 32 + ln) * HDIM + s * 32 + hl * 8;
        k0.h[0] = *(const bf16x8*)p0;  k0.h[1] = *(const bf16x8*)(p0 + 16);
        const unsigned short* p1 = Kp + (size_t)(c * 32 + 16 + ln) * HDIM + s * 32 + hl * 8;
        k1.h[0] = *(const bf16x8*)p1;  k1.h[1] = *(const bf16x8*)(p1 + 16);
        s0 = __builtin_amdgcn_wmma_f32_16x16x32_bf16(false, qf[s].f, false, k0.f, (short)0, s0, false, false);
        s1 = __builtin_amdgcn_wmma_f32_16x16x32_bf16(false, qf[s].f, false, k1.f, (short)0, s1, false, false);
      }

      const int key0 = c * 32 + ln, key1 = key0 + 16;
#pragma unroll
      for (int e = 0; e < 8; ++e) {   // online softmax, row = e + 8*hl
        const int qr = qbase + w * 16 + e + 8 * hl;
        float v0 = (key0 <= qr) ? s0[e] : -1e30f;
        float v1 = (key1 <= qr) ? s1[e] : -1e30f;
        float mx = fmaxf(v0, v1);
        mx = fmaxf(mx, __shfl_xor(mx, 1, 16));
        mx = fmaxf(mx, __shfl_xor(mx, 2, 16));
        mx = fmaxf(mx, __shfl_xor(mx, 4, 16));
        mx = fmaxf(mx, __shfl_xor(mx, 8, 16));
        const float mnew = fmaxf(mrow[e], mx);
        const float fs = __expf(mrow[e] - mnew);
        const float p0 = __expf(v0 - mnew);
        const float p1 = __expf(v1 - mnew);
        float rs = p0 + p1;
        rs += __shfl_xor(rs, 1, 16);
        rs += __shfl_xor(rs, 2, 16);
        rs += __shfl_xor(rs, 4, 16);
        rs += __shfl_xor(rs, 8, 16);
        lrow[e] = lrow[e] * fs + rs;
        mrow[e] = mnew;
#pragma unroll
        for (int c4 = 0; c4 < 4; ++c4) O[c4][e] = O[c4][e] * fs;
        const int prow = e + 8 * hl;
        Ps[w * 512 + prow * 32 + ln]      = f2bf(p0);
        Ps[w * 512 + prow * 32 + 16 + ln] = f2bf(p1);
      }

      BFrag pf;  // reload P in A-layout (same-wave LDS round trip)
      pf.h[0] = *(const bf16x8*)(Ps + w * 512 + ln * 32 + 8 * hl);
      pf.h[1] = *(const bf16x8*)(Ps + w * 512 + ln * 32 + 8 * hl + 16);
      const unsigned short* Vb = Vs + cur * 2048;
#pragma unroll
      for (int c4 = 0; c4 < 4; ++c4) {
        BFrag vf;
        const unsigned short* vp = Vb + (c4 * 16 + ln) * 32 + hl * 8;
        vf.h[0] = *(const bf16x8*)vp;
        vf.h[1] = *(const bf16x8*)(vp + 16);
        O[c4] = __builtin_amdgcn_wmma_f32_16x16x32_bf16(
            false, pf.f, false, vf.f, (short)0, O[c4], false, false);
      }
    }
  }

#pragma unroll
  for (int e = 0; e < 8; ++e) {
    const float rcp = 1.0f / lrow[e];
    const int q = qbase + w * 16 + e + 8 * hl;
#pragma unroll
    for (int c4 = 0; c4 < 4; ++c4) {
      const int col = h * HDIM + c4 * 16 + ln;
      ctx[(size_t)(b * NSEQ + q) * DOUT + col] = f2bf(O[c4][e] * rcp);
    }
  }
}

// ---------------------------------------------------------------- out proj
__global__ void __launch_bounds__(256, 1) out_gemm(
    const unsigned short* __restrict__ ctx,
    const unsigned short* __restrict__ wo,   // out_w row-major == [n][k]
    const float* __restrict__ bias,
    float* __restrict__ out) {
  __shared__ unsigned short As[2 * 128 * 32];
  __shared__ unsigned short Bs[2 * 128 * 32];

  const int nbase = blockIdx.x * 128, mbase = blockIdx.y * 128;
  f32x8 acc[8] = {};
  gemm_mainloop_bf16(ctx, wo, mbase, nbase, As, Bs, acc);

  const int t = threadIdx.x;
  const int w = t >> 5, lane = t & 31, hl = lane >> 4, ln = lane & 15;
#pragma unroll
  for (int c = 0; c < 8; ++c) {
    const int col = nbase + c * 16 + ln;
    const float bv = bias[col];
#pragma unroll
    for (int e = 0; e < 8; ++e) {
      const int m = mbase + w * 16 + e + 8 * hl;
      out[(size_t)m * DOUT + col] = acc[c][e] + bv;
    }
  }
}

// ---------------------------------------------------------------- launch
extern "C" void kernel_launch(void* const* d_in, const int* in_sizes, int n_in,
                              void* d_out, int out_size, void* d_ws, size_t ws_size,
                              hipStream_t stream) {
  const float* x   = (const float*)d_in[0];
  const float* w_q = (const float*)d_in[1];
  const float* w_k = (const float*)d_in[2];
  const float* w_v = (const float*)d_in[3];
  const float* w_o = (const float*)d_in[4];
  const float* o_b = (const float*)d_in[5];
  float* out = (float*)d_out;

  unsigned short* ws = (unsigned short*)d_ws;
  unsigned short* xb  = ws;                 // 8M elems; reused as ctx later
  unsigned short* wqb = ws + 8388608;       // transposed [n][k]
  unsigned short* wkb = ws + 9437184;       // transposed [n][k]
  unsigned short* wvb = ws + 10485760;      // transposed [n][k]
  unsigned short* wob = ws + 11534336;      // row-major (already [n][k])
  unsigned short* Qb  = ws + 12582912;      // [b][h][n][d] bf16
  unsigned short* Kb  = ws + 20971520;      // [b][h][n][d] bf16
  unsigned short* Vb  = ws + 29360128;      // [b][h][d][n] bf16

  cvt_bf16  <<<dim3(8192), dim3(256), 0, stream>>>(x,   xb,  8388608);
  cvt_bf16  <<<dim3(1024), dim3(256), 0, stream>>>(w_o, wob, 1048576);
  cvt_bf16_t<<<dim3(32, 32), dim3(32, 8), 0, stream>>>(w_q, wqb);
  cvt_bf16_t<<<dim3(32, 32), dim3(32, 8), 0, stream>>>(w_k, wkb);
  cvt_bf16_t<<<dim3(32, 32), dim3(32, 8), 0, stream>>>(w_v, wvb);

  qkv_gemm   <<<dim3(8, 64, 3), dim3(256), 0, stream>>>(xb, wqb, wkb, wvb, Qb, Kb, Vb);
  attn_kernel<<<dim3(16, 64),   dim3(256), 0, stream>>>(Qb, Kb, Vb, xb /*ctx*/);
  out_gemm   <<<dim3(8, 64),    dim3(256), 0, stream>>>(xb, wob, o_b, out);
}
